// MovieRecommender_45406394253580
// MI455X (gfx1250) — matrix operand order repaired and checked by reading, not compiled, not run on hardware
//
#include <hip/hip_runtime.h>
#include <hip/hip_bf16.h>
#include <math.h>

// ---------------------------------------------------------------------------
// MovieRecommender on MI455X (gfx1250, wave32, WMMA)
// Strategy: item tower is a function of movie id only -> compute once per
// movie (20001 rows) with v_wmma_f32_16x16x32_bf16, then pooling is a cheap
// gather from the L2-resident table, then user MLP + dot via WMMA.
// ---------------------------------------------------------------------------

typedef __attribute__((ext_vector_type(16))) __bf16 bf16x16;
typedef __attribute__((ext_vector_type(8)))  __bf16 bf16x8;
typedef __attribute__((ext_vector_type(8)))  float  f32x8;

#define PADIDX   20000          // MOVIES; padding row index
#define NMOVIES  20001          // MOVIES + 1 (incl. pad row)
#define NB       1024
#define NHIST    50
#define ODIM     128
#define PHID     256

// ---- workspace layout (bytes, 256-aligned) --------------------------------
static constexpr size_t SZ_ITEM  = (size_t)NMOVIES * 128 * 4;   // item tower out, f32
static constexpr size_t SZ_GENT  = (size_t)NMOVIES * 16  * 4;   // genome tower out, f32
static constexpr size_t SZ_WTAG  = 16  * 1024 * 2;              // w_tag^T  bf16 (16 x 1024)
static constexpr size_t SZ_WGEN  = 16  * 1152 * 2;              // w_genome^T (16 x 1152)
static constexpr size_t SZ_WIP1  = 256 * 96   * 2;              // w_ip1^T (256 x 96)
static constexpr size_t SZ_WIP2  = 128 * 256  * 2;              // w_ip2^T (128 x 256)
static constexpr size_t SZ_WUP1  = 256 * 192  * 2;              // w_up1^T (256 x 192)
static constexpr size_t SZ_WUP2  = 128 * 256  * 2;              // w_up2^T (128 x 256)
static constexpr size_t SZ_UCAT  = (size_t)NB * 192 * 2;        // user_cat bf16 (1024 x 192)

static constexpr size_t alignup(size_t x) { return (x + 255) & ~(size_t)255; }
static constexpr size_t OFF_ITEM = 0;
static constexpr size_t OFF_GENT = alignup(OFF_ITEM + SZ_ITEM);
static constexpr size_t OFF_WTAG = alignup(OFF_GENT + SZ_GENT);
static constexpr size_t OFF_WGEN = alignup(OFF_WTAG + SZ_WTAG);
static constexpr size_t OFF_WIP1 = alignup(OFF_WGEN + SZ_WGEN);
static constexpr size_t OFF_WIP2 = alignup(OFF_WIP1 + SZ_WIP1);
static constexpr size_t OFF_WUP1 = alignup(OFF_WIP2 + SZ_WIP2);
static constexpr size_t OFF_WUP2 = alignup(OFF_WUP1 + SZ_WUP1);
static constexpr size_t OFF_UCAT = alignup(OFF_WUP2 + SZ_WUP2);

// ---- WMMA fragment helpers -------------------------------------------------
__device__ __forceinline__ bf16x16 make_frag(bf16x8 lo, bf16x8 hi) {
    bf16x16 f;
#pragma unroll
    for (int i = 0; i < 8; ++i) { f[i] = lo[i]; f[i + 8] = hi[i]; }
    return f;
}

// A fragment (16x32 bf16, M x K): lanes 0-15 row M=lane hold K runs {k0..k0+7},
// {k0+16..k0+23}; lanes 16-31 hold {k0+8..15},{k0+24..31}  (ISA 7.12.2)
__device__ __forceinline__ bf16x16 load_a_lds(const __bf16* base, int stride, int k0) {
    int lane = threadIdx.x & 31;
    const __bf16* p = base + (lane & 15) * stride + k0 + (lane >> 4) * 8;
    return make_frag(*(const bf16x8*)p, *(const bf16x8*)(p + 16));
}
__device__ __forceinline__ bf16x16 load_a_global(const __bf16* base, int stride,
                                                 int row0, int k0) {
    int lane = threadIdx.x & 31;
    const __bf16* p = base + (size_t)(row0 + (lane & 15)) * stride + k0 + (lane >> 4) * 8;
    return make_frag(*(const bf16x8*)p, *(const bf16x8*)(p + 16));
}
// B fragment (32x16 bf16, K x N), weights stored transposed N x Kpad:
// lanes 0-15 col N=lane hold K runs {k0..k0+7},{k0+16..k0+23}; lanes 16-31 {k0+8..15},{k0+24..31}.
__device__ __forceinline__ bf16x16 load_b_global(const __bf16* w, int stride,
                                                 int n0, int k0) {
    int lane = threadIdx.x & 31;
    const __bf16* p = w + (size_t)(n0 + (lane & 15)) * stride + k0 + (lane >> 4) * 8;
    return make_frag(*(const bf16x8*)p, *(const bf16x8*)(p + 16));
}
__device__ __forceinline__ f32x8 wmma_bf16(bf16x16 a, bf16x16 b, f32x8 c) {
    return __builtin_amdgcn_wmma_f32_16x16x32_bf16(false, a, false, b, (short)0, c,
                                                   false, false);
}

// ---- weight transpose + f32->bf16 pack -------------------------------------
__global__ void transpose_pad_bf16(__bf16* __restrict__ dst, const float* __restrict__ src,
                                   int K, int N, int Kpad) {
    int idx = blockIdx.x * 256 + threadIdx.x;
    if (idx >= N * Kpad) return;
    int n = idx / Kpad, k = idx - n * Kpad;
    dst[(size_t)n * Kpad + k] = (k < K) ? (__bf16)src[(size_t)k * N + n] : (__bf16)0.0f;
}

// big tower accumulation: (16 x Klen) f32 ctx tile @ (Klen x 16) weights
__device__ f32x8 big_tower(const float* __restrict__ ctx, int Klen, int Kpad,
                           const __bf16* __restrict__ wT, __bf16* Abuf, int m0, int tid) {
    f32x8 acc = {0.f, 0.f, 0.f, 0.f, 0.f, 0.f, 0.f, 0.f};
    for (int k0 = 0; k0 < Kpad; k0 += 256) {
        int cw = Kpad - k0; if (cw > 256) cw = 256;
        __syncthreads();                    // prior consumers of Abuf done
        int cv = cw >> 2;                   // float4s per row
        for (int idx = tid; idx < 16 * cv; idx += 32) {
            int r = idx / cv, c4 = (idx - r * cv) << 2;
            int k = k0 + c4;
            int m = m0 + r; if (m > PADIDX) m = PADIDX;
            float4 v = make_float4(0.f, 0.f, 0.f, 0.f);
            if (k < Klen) v = *(const float4*)(ctx + (size_t)m * Klen + k);
            __bf16* d = Abuf + r * 256 + c4;
            d[0] = (__bf16)v.x; d[1] = (__bf16)v.y; d[2] = (__bf16)v.z; d[3] = (__bf16)v.w;
        }
        __syncthreads();
        int ksteps = cw >> 5;
        for (int ks = 0; ks < ksteps; ++ks) {
            bf16x16 a = load_a_lds(Abuf, 256, ks * 32);
            bf16x16 b = load_b_global(wT, Kpad, 0, k0 + ks * 32);
            acc = wmma_bf16(a, b, acc);
        }
    }
    return acc;
}

// ---- phase 1: item tower for every movie (one wave == one 16-movie tile) ---
__global__ __launch_bounds__(32)
void item_tower_kernel(const float* __restrict__ genre_ctx, const float* __restrict__ tag_ctx,
                       const float* __restrict__ genome_ctx, const int* __restrict__ year_ctx,
                       const float* __restrict__ item_emb, const float* __restrict__ year_tab,
                       const float* __restrict__ w_genre, const float* __restrict__ b_genre,
                       const float* __restrict__ w_item,  const float* __restrict__ b_item,
                       const float* __restrict__ w_year,  const float* __restrict__ b_year,
                       const float* __restrict__ b_tag,   const float* __restrict__ b_genome,
                       const float* __restrict__ b_ip1,   const float* __restrict__ b_ip2,
                       const __bf16* __restrict__ wTtag,  const __bf16* __restrict__ wTgen,
                       const __bf16* __restrict__ wTip1,  const __bf16* __restrict__ wTip2,
                       float* __restrict__ itemEmbOut, float* __restrict__ genomeOut) {
    __shared__ __bf16 Abuf[16 * 256];
    __shared__ __bf16 catb[16 * 96];
    __shared__ __bf16 hbuf[16 * 256];
    __shared__ float  gbuf[16 * 20];
    __shared__ float  ibuf[16 * 32];
    __shared__ float  ybuf[16 * 8];
    __shared__ int    yid[16];

    const int tid = threadIdx.x;
    const int m0  = blockIdx.x * 16;
    const int lane = tid, col = lane & 15, rbase = (lane >> 4) * 8;

    // -- tag tower (K=1000, pad 1024) -> catb[:, 8:24]
    f32x8 accT = big_tower(tag_ctx, 1000, 1024, wTtag, Abuf, m0, tid);
#pragma unroll
    for (int v = 0; v < 8; ++v)
        catb[(rbase + v) * 96 + 8 + col] = (__bf16)tanhf(accT[v] + b_tag[col]);

    // -- genome tower (K=1128, pad 1152) -> catb[:, 24:40] + genomeOut table
    f32x8 accG = big_tower(genome_ctx, 1128, 1152, wTgen, Abuf, m0, tid);
#pragma unroll
    for (int v = 0; v < 8; ++v) {
        float x = tanhf(accG[v] + b_genome[col]);
        catb[(rbase + v) * 96 + 24 + col] = (__bf16)x;
        int m = m0 + rbase + v; if (m > PADIDX) m = PADIDX;
        genomeOut[(size_t)m * 16 + col] = x;
    }

    // -- small towers (VALU): genre(20->8), item(32->32), year(8->8)
    for (int idx = tid; idx < 16 * 20; idx += 32) {
        int r = idx / 20, c = idx - r * 20;
        int m = m0 + r; if (m > PADIDX) m = PADIDX;
        gbuf[idx] = genre_ctx[(size_t)m * 20 + c];
    }
    for (int idx = tid; idx < 16 * 32; idx += 32) {
        int r = idx >> 5, c = idx & 31;
        int m = m0 + r; if (m > PADIDX) m = PADIDX;
        ibuf[idx] = item_emb[(size_t)m * 32 + c];
    }
    if (tid < 16) { int m = m0 + tid; if (m > PADIDX) m = PADIDX; yid[tid] = year_ctx[m]; }
    __syncthreads();
    for (int idx = tid; idx < 16 * 8; idx += 32)
        ybuf[idx] = year_tab[yid[idx >> 3] * 8 + (idx & 7)];
    __syncthreads();
    for (int o = tid; o < 16 * 8; o += 32) {               // genre -> catb[:, 0:8]
        int r = o >> 3, j = o & 7; float s = b_genre[j];
        for (int i = 0; i < 20; ++i) s += gbuf[r * 20 + i] * w_genre[i * 8 + j];
        catb[r * 96 + j] = (__bf16)tanhf(s);
    }
    for (int o = tid; o < 16 * 32; o += 32) {              // item -> catb[:, 40:72]
        int r = o >> 5, j = o & 31; float s = b_item[j];
        for (int i = 0; i < 32; ++i) s += ibuf[r * 32 + i] * w_item[i * 32 + j];
        catb[r * 96 + 40 + j] = (__bf16)tanhf(s);
    }
    for (int o = tid; o < 16 * 8; o += 32) {               // year -> catb[:, 72:80]
        int r = o >> 3, j = o & 7; float s = b_year[j];
        for (int i = 0; i < 8; ++i) s += ybuf[r * 8 + i] * w_year[i * 8 + j];
        catb[r * 96 + 72 + j] = (__bf16)tanhf(s);
    }
    for (int o = tid; o < 16 * 16; o += 32)                // pad cols 80..95
        catb[(o >> 4) * 96 + 80 + (o & 15)] = (__bf16)0.0f;
    __syncthreads();

    // -- MLP1: (16x96) @ (96x256), relu -> hbuf
    for (int nt = 0; nt < 16; ++nt) {
        f32x8 acc = {0.f, 0.f, 0.f, 0.f, 0.f, 0.f, 0.f, 0.f};
#pragma unroll
        for (int ks = 0; ks < 3; ++ks) {
            bf16x16 a = load_a_lds(catb, 96, ks * 32);
            bf16x16 b = load_b_global(wTip1, 96, nt * 16, ks * 32);
            acc = wmma_bf16(a, b, acc);
        }
#pragma unroll
        for (int v = 0; v < 8; ++v) {
            float x = acc[v] + b_ip1[nt * 16 + col];
            hbuf[(rbase + v) * 256 + nt * 16 + col] = (__bf16)(x > 0.f ? x : 0.f);
        }
    }
    __syncthreads();

    // -- MLP2: (16x256) @ (256x128) -> item table (f32)
    for (int nt = 0; nt < 8; ++nt) {
        f32x8 acc = {0.f, 0.f, 0.f, 0.f, 0.f, 0.f, 0.f, 0.f};
#pragma unroll
        for (int ks = 0; ks < 8; ++ks) {
            bf16x16 a = load_a_lds(hbuf, 256, ks * 32);
            bf16x16 b = load_b_global(wTip2, 256, nt * 16, ks * 32);
            acc = wmma_bf16(a, b, acc);
        }
#pragma unroll
        for (int v = 0; v < 8; ++v) {
            int m = m0 + rbase + v; if (m > PADIDX) m = PADIDX;
            itemEmbOut[(size_t)m * 128 + nt * 16 + col] = acc[v] + b_ip2[nt * 16 + col];
        }
    }
}

// ---- phase 2a: rating-weighted pooling + small user towers -> ucat bf16 ----
__global__ __launch_bounds__(128)
void user_pool_kernel(const int* __restrict__ hist, const float* __restrict__ ratings,
                      const float* __restrict__ ugc, const int* __restrict__ timestamps,
                      const float* __restrict__ itemEmbT, const float* __restrict__ genomeT,
                      const float* __restrict__ w_ugenre, const float* __restrict__ b_ugenre,
                      const float* __restrict__ w_ts, const float* __restrict__ b_ts,
                      const float* __restrict__ ts_tab, __bf16* __restrict__ ucat) {
    int b = blockIdx.x, tid = threadIdx.x;
    float wsum = 0.f, acc = 0.f, gacc = 0.f;
    for (int h = 0; h < NHIST; ++h) {
        int id  = hist[b * NHIST + h];
        float r = ratings[b * NHIST + h];
        float w = (id == PADIDX) ? 0.f : r;
        wsum += fabsf(w);
        acc  += w * itemEmbT[(size_t)id * 128 + tid];
        if (tid < 16) gacc += w * genomeT[(size_t)id * 16 + tid];
    }
    wsum = fmaxf(wsum, 1e-6f);
    __bf16* dst = ucat + (size_t)b * 192;
    dst[tid] = (__bf16)(acc / wsum);                       // item_pool [0:128)
    if (tid < 16) dst[128 + tid] = (__bf16)(gacc / wsum);  // genome_emb [128:144)
    if (tid < 32) {                                        // genre_emb [144:176)
        float s = b_ugenre[tid];
        for (int i = 0; i < 20; ++i) s += ugc[b * 20 + i] * w_ugenre[i * 32 + tid];
        dst[144 + tid] = (__bf16)tanhf(s);
    }
    if (tid < 4) {                                         // ts_emb [176:180)
        int t = timestamps[b];
        float s = b_ts[tid];
        for (int i = 0; i < 4; ++i) s += ts_tab[t * 4 + i] * w_ts[i * 4 + tid];
        dst[176 + tid] = (__bf16)tanhf(s);
    }
    if (tid < 12) dst[180 + tid] = (__bf16)0.0f;           // pad [180:192)
}

// ---- phase 2b: user MLP (WMMA) + final dot ---------------------------------
__global__ __launch_bounds__(32)
void user_mlp_dot_kernel(const __bf16* __restrict__ ucat,
                         const __bf16* __restrict__ wTup1, const __bf16* __restrict__ wTup2,
                         const float* __restrict__ b_up1, const float* __restrict__ b_up2,
                         const float* __restrict__ itemEmbT, const int* __restrict__ target,
                         float* __restrict__ out) {
    __shared__ __bf16 hb[16 * 256];
    __shared__ float  fb[16 * 128];
    const int tid = threadIdx.x, u0 = blockIdx.x * 16;
    const int col = tid & 15, rbase = (tid >> 4) * 8;

    for (int nt = 0; nt < 16; ++nt) {                      // (16x192)@(192x256), relu
        f32x8 acc = {0.f, 0.f, 0.f, 0.f, 0.f, 0.f, 0.f, 0.f};
#pragma unroll
        for (int ks = 0; ks < 6; ++ks) {
            bf16x16 a = load_a_global(ucat, 192, u0, ks * 32);
            bf16x16 b = load_b_global(wTup1, 192, nt * 16, ks * 32);
            acc = wmma_bf16(a, b, acc);
        }
#pragma unroll
        for (int v = 0; v < 8; ++v) {
            float x = acc[v] + b_up1[nt * 16 + col];
            hb[(rbase + v) * 256 + nt * 16 + col] = (__bf16)(x > 0.f ? x : 0.f);
        }
    }
    __syncthreads();
    for (int nt = 0; nt < 8; ++nt) {                       // (16x256)@(256x128)
        f32x8 acc = {0.f, 0.f, 0.f, 0.f, 0.f, 0.f, 0.f, 0.f};
#pragma unroll
        for (int ks = 0; ks < 8; ++ks) {
            bf16x16 a = load_a_lds(hb, 256, ks * 32);
            bf16x16 b = load_b_global(wTup2, 256, nt * 16, ks * 32);
            acc = wmma_bf16(a, b, acc);
        }
#pragma unroll
        for (int v = 0; v < 8; ++v)
            fb[(rbase + v) * 128 + nt * 16 + col] = acc[v] + b_up2[nt * 16 + col];
    }
    __syncthreads();
    if (tid < 16) {                                        // score = <user, item[target]>
        int b  = u0 + tid;
        const float* ir = itemEmbT + (size_t)target[b] * 128;
        float s = 0.f;
        for (int c = 0; c < 128; ++c) s += fb[tid * 128 + c] * ir[c];
        out[b] = s;
    }
}

// ---------------------------------------------------------------------------
extern "C" void kernel_launch(void* const* d_in, const int* in_sizes, int n_in,
                              void* d_out, int out_size, void* d_ws, size_t ws_size,
                              hipStream_t stream) {
    const float* ugc        = (const float*)d_in[0];
    const int*   hist       = (const int*)  d_in[1];
    const float* ratings    = (const float*)d_in[2];
    const int*   timestamps = (const int*)  d_in[3];
    const int*   target     = (const int*)  d_in[4];
    const float* genome_ctx = (const float*)d_in[5];
    const float* genre_ctx  = (const float*)d_in[6];
    const float* tag_ctx    = (const float*)d_in[7];
    const int*   year_ctx   = (const int*)  d_in[8];
    const float* item_emb   = (const float*)d_in[9];
    const float* year_tab   = (const float*)d_in[10];
    const float* ts_tab     = (const float*)d_in[11];
    const float* w_item   = (const float*)d_in[12]; const float* b_item   = (const float*)d_in[13];
    const float* w_genre  = (const float*)d_in[14]; const float* b_genre  = (const float*)d_in[15];
    const float* w_tag    = (const float*)d_in[16]; const float* b_tag    = (const float*)d_in[17];
    const float* w_genome = (const float*)d_in[18]; const float* b_genome = (const float*)d_in[19];
    const float* w_year   = (const float*)d_in[20]; const float* b_year   = (const float*)d_in[21];
    const float* w_ugenre = (const float*)d_in[22]; const float* b_ugenre = (const float*)d_in[23];
    const float* w_ts     = (const float*)d_in[24]; const float* b_ts     = (const float*)d_in[25];
    const float* w_up1    = (const float*)d_in[26]; const float* b_up1    = (const float*)d_in[27];
    const float* w_up2    = (const float*)d_in[28]; const float* b_up2    = (const float*)d_in[29];
    const float* w_ip1    = (const float*)d_in[30]; const float* b_ip1    = (const float*)d_in[31];
    const float* w_ip2    = (const float*)d_in[32]; const float* b_ip2    = (const float*)d_in[33];

    char* ws = (char*)d_ws;
    float*  itemEmbT = (float*) (ws + OFF_ITEM);
    float*  genomeT  = (float*) (ws + OFF_GENT);
    __bf16* wTtag    = (__bf16*)(ws + OFF_WTAG);
    __bf16* wTgen    = (__bf16*)(ws + OFF_WGEN);
    __bf16* wTip1    = (__bf16*)(ws + OFF_WIP1);
    __bf16* wTip2    = (__bf16*)(ws + OFF_WIP2);
    __bf16* wTup1    = (__bf16*)(ws + OFF_WUP1);
    __bf16* wTup2    = (__bf16*)(ws + OFF_WUP2);
    __bf16* ucat     = (__bf16*)(ws + OFF_UCAT);

    auto tp = [&](__bf16* dst, const float* src, int K, int N, int Kpad) {
        int n = N * Kpad;
        transpose_pad_bf16<<<(n + 255) / 256, 256, 0, stream>>>(dst, src, K, N, Kpad);
    };
    tp(wTtag, w_tag,    1000, 16,  1024);
    tp(wTgen, w_genome, 1128, 16,  1152);
    tp(wTip1, w_ip1,    80,   256, 96);
    tp(wTip2, w_ip2,    256,  128, 256);
    tp(wTup1, w_up1,    180,  256, 192);
    tp(wTup2, w_up2,    256,  128, 256);

    item_tower_kernel<<<(NMOVIES + 15) / 16, 32, 0, stream>>>(
        genre_ctx, tag_ctx, genome_ctx, year_ctx, item_emb, year_tab,
        w_genre, b_genre, w_item, b_item, w_year, b_year,
        b_tag, b_genome, b_ip1, b_ip2,
        wTtag, wTgen, wTip1, wTip2, itemEmbT, genomeT);

    user_pool_kernel<<<NB, 128, 0, stream>>>(
        hist, ratings, ugc, timestamps, itemEmbT, genomeT,
        w_ugenre, b_ugenre, w_ts, b_ts, ts_tab, ucat);

    user_mlp_dot_kernel<<<NB / 16, 32, 0, stream>>>(
        ucat, wTup1, wTup2, b_up1, b_up2, itemEmbT, target, (float*)d_out);
}
